// GroupedQueryAttention_10771777979068
// MI455X (gfx1250) — matrix operand order repaired and checked by reading, not compiled, not run on hardware
//
#include <hip/hip_runtime.h>
#include <hip/hip_bf16.h>
#include <math.h>

// ---- problem constants (from reference) ----
#define B_    2
#define S_    4096
#define DIN   640
#define H_    4
#define HD_   256
#define DOUT  (H_ * HD_)          // 1024
static constexpr float SCALING_F = 0.0625f;   // 256^-0.5
static constexpr float EPS_F     = 1e-6f;

typedef _Float16 half16 __attribute__((ext_vector_type(16)));
typedef float    float8 __attribute__((ext_vector_type(8)));

// ---------------------------------------------------------------------------
// CDNA5 primitives
// ---------------------------------------------------------------------------
__device__ __forceinline__ float8 wmma_f16(half16 a, half16 b, float8 c) {
  return __builtin_amdgcn_wmma_f32_16x16x32_f16(false, a, false, b,
                                                (short)0, c, false, false);
}

// Async global->LDS copy, 16 bytes per lane (ASYNCcnt-tracked).
__device__ __forceinline__ void async_b128(uint32_t lds_off, const void* g) {
  asm volatile("global_load_async_to_lds_b128 %0, %1, off"
               :: "v"(lds_off), "v"(g) : "memory");
}
__device__ __forceinline__ void wait_async0() {
  asm volatile("s_wait_asynccnt 0" ::: "memory");
}
__device__ __forceinline__ uint32_t lds_addr(const void* p) {
  return (uint32_t)(size_t)p;   // generic shared ptr: low 32 bits = LDS offset
}

// A-fragment (16x32 f16), row-major source: elem(m,k)=src[(row0+m)*ld + kb + k]
// M = lane%16; lanes<16 hold K {0..7,16..23}, lanes>=16 hold K {8..15,24..31}.
__device__ __forceinline__ half16 frag_a(const _Float16* __restrict__ src,
                                         int row0, int ld, int kb, int lane) {
  half16 f;
  const int m    = lane & 15;
  const int koff = (lane < 16) ? 0 : 8;
  const _Float16* p = src + (size_t)(row0 + m) * ld + kb;
#pragma unroll
  for (int j = 0; j < 4; ++j)
    ((uint32_t*)&f)[j] = *(const uint32_t*)(p + koff + 2 * j);
#pragma unroll
  for (int j = 0; j < 4; ++j)
    ((uint32_t*)&f)[4 + j] = *(const uint32_t*)(p + 16 + koff + 2 * j);
  return f;
}

// B-fragment (32x16 f16) where B[k][n] = src[(n0+n)*ld + kb + k]
// N = lane%16; lanes<16 hold K {0..7,16..23}, lanes>=16 hold K {8..15,24..31}.
__device__ __forceinline__ half16 frag_bT(const _Float16* __restrict__ src,
                                          int n0, int ld, int kb, int lane) {
  half16 f;
  const int n    = lane & 15;
  const int koff = (lane < 16) ? 0 : 8;
  const _Float16* p = src + (size_t)(n0 + n) * ld + kb + koff;
#pragma unroll
  for (int j = 0; j < 8; ++j)
    ((uint32_t*)&f)[j] = *(const uint32_t*)(p + (j >> 2) * 16 + 2 * (j & 3));
  return f;
}

// ---------------------------------------------------------------------------
// Kernel 0: one-shot f32 -> f16 weight conversion (removes per-tile cvt work)
// ---------------------------------------------------------------------------
__global__ __launch_bounds__(256)
void cvt_f32_f16_kernel(const float* __restrict__ src, _Float16* __restrict__ dst,
                        int npairs) {
  int i = blockIdx.x * 256 + threadIdx.x;
  if (i < npairs) {
    float2 w = ((const float2*)src)[i];
    dst[2 * i]     = (_Float16)w.x;
    dst[2 * i + 1] = (_Float16)w.y;
  }
}

// ---------------------------------------------------------------------------
// Kernel 1: QKV projection + RMS norm + RoPE (+ q scaling folded in)
// grid: (B*S/16, 6) — y: 0..3 = q heads, 4 = k, 5 = v.  block: 128 (4 waves)
// ---------------------------------------------------------------------------
__global__ __launch_bounds__(128)
void qkv_norm_rope_kernel(const float* __restrict__ x,
                          const _Float16* __restrict__ Wq16,
                          const _Float16* __restrict__ Wk16,
                          const _Float16* __restrict__ Wv16,
                          const float* __restrict__ q_scale,
                          const float* __restrict__ k_scale,
                          const float* __restrict__ cosd,
                          const float* __restrict__ sind,
                          _Float16* __restrict__ qf,
                          _Float16* __restrict__ kf,
                          _Float16* __restrict__ vf) {
  __shared__ _Float16 xs[16 * DIN];     // 20 KB  x tile in f16
  __shared__ float    acc[16 * HD_];    // 16 KB  f32 head outputs
  __shared__ float    red[16 * 8];
  __shared__ _Float16 os[16 * HD_];     // 8 KB   f16 staging for vector store

  const int tile = blockIdx.x;
  const int head = blockIdx.y;                 // 0..3 q, 4 k, 5 v
  const int b    = tile / (S_ / 16);
  const int s0   = (tile % (S_ / 16)) * 16;
  const int tid  = threadIdx.x;
  const int wave = tid >> 5, lane = tid & 31;

  // stage x tile (16 tokens x 640) as f16, pairwise
  {
    const float2* xrow = (const float2*)(x + ((size_t)b * S_ + s0) * DIN);
    uint32_t* dst = (uint32_t*)xs;
    for (int i = tid; i < 16 * DIN / 2; i += 128) {
      float2 w = xrow[i];
      _Float16 lo = (_Float16)w.x, hi = (_Float16)w.y;
      dst[i] = (uint32_t)__builtin_bit_cast(uint16_t, lo) |
               ((uint32_t)__builtin_bit_cast(uint16_t, hi) << 16);
    }
  }
  __syncthreads();

  const _Float16* W;
  int orow0;
  if (head < H_)        { W = Wq16; orow0 = head * HD_; }
  else if (head == H_)  { W = Wk16; orow0 = 0; }
  else                  { W = Wv16; orow0 = 0; }

  float8 c[4] = {};
  for (int kb = 0; kb < DIN; kb += 32) {
    half16 a = frag_a(xs, 0, DIN, kb, lane);
#pragma unroll
    for (int t = 0; t < 4; ++t)
      c[t] = wmma_f16(a, frag_bT(W, orow0 + wave * 64 + t * 16, DIN, kb, lane), c[t]);
  }

  // spill accumulators to LDS: acc[token][col]
  const int nlo = lane & 15, hh = lane >> 4;
#pragma unroll
  for (int t = 0; t < 4; ++t)
#pragma unroll
    for (int j = 0; j < 8; ++j)
      acc[(j + 8 * hh) * HD_ + wave * 64 + t * 16 + nlo] = c[t][j];
  __syncthreads();

  const bool is_v = (head == 5);
  const bool is_q = (head < H_);

  // RMS-norm sum of squares: 8 threads per token, 32 elements each
  const int tok = tid >> 3, sub = tid & 7;
  if (!is_v) {
    float ss = 0.f;
#pragma unroll
    for (int i = sub * 32; i < sub * 32 + 32; ++i) {
      float v = acc[tok * HD_ + i];
      ss += v * v;
    }
    red[tok * 8 + sub] = ss;
  }
  __syncthreads();

  float rn = 1.f;
  if (!is_v) {
    float tot = 0.f;
#pragma unroll
    for (int i = 0; i < 8; ++i) tot += red[tok * 8 + i];
    rn = rsqrtf(tot * (1.0f / HD_) + EPS_F);
  }

  const float* scl = is_q ? q_scale : k_scale;
  const int s_abs = s0 + tok;
  for (int i = sub * 32; i < sub * 32 + 32; ++i) {
    float outv;
    if (is_v) {
      outv = acc[tok * HD_ + i];
    } else {
      float xn = acc[tok * HD_ + i] * rn * (1.f + scl[i]);
      int ip = (i < HD_ / 2) ? (i + HD_ / 2) : (i - HD_ / 2);
      float xp = acc[tok * HD_ + ip] * rn * (1.f + scl[ip]);
      float rot = (i < HD_ / 2) ? -xp : xp;
      float cv = cosd[(size_t)s_abs * HD_ + i];
      float sv = sind[(size_t)s_abs * HD_ + i];
      outv = xn * cv + rot * sv;
      if (is_q) outv *= SCALING_F;
    }
    os[tok * HD_ + i] = (_Float16)outv;
  }
  __syncthreads();

  // vectorized contiguous store of the whole 16x256 block
  _Float16* dst;
  if (is_q)       dst = qf + (((size_t)b * H_ + head) * S_ + s0) * HD_;
  else if (!is_v) dst = kf + ((size_t)b * S_ + s0) * HD_;
  else            dst = vf + ((size_t)b * S_ + s0) * HD_;
  {
    const uint4* s = (const uint4*)os;
    uint4* d = (uint4*)dst;
    for (int i = tid; i < 16 * HD_ / 8; i += 128) d[i] = s[i];
  }
}

// ---------------------------------------------------------------------------
// Kernel 2: causal flash attention. grid: (S/64, B*H), block: 128 (4 waves).
// K staged via global_load_async_to_lds_b128 (ASYNCcnt); V transposed manually.
// ---------------------------------------------------------------------------
__global__ __launch_bounds__(128)
void flash_attn_kernel(const _Float16* __restrict__ qf,
                       const _Float16* __restrict__ kf,
                       const _Float16* __restrict__ vf,
                       _Float16* __restrict__ ctxf) {
  __shared__ _Float16 Qs[64 * HD_];       // 32 KB  queries (reused for ctx out)
  __shared__ _Float16 Ks[32 * HD_];       // 16 KB  [key][d] row-major
  __shared__ _Float16 Vt[HD_ * 32];       // 16 KB  [d][key] transposed
  __shared__ _Float16 ps[4][16 * 32];     // 4 KB   per-wave P tile (f16)

  const int bh   = blockIdx.y;
  const int b    = bh / H_, h = bh % H_;
  const int qblk = blockIdx.x;
  const int tid  = threadIdx.x, wave = tid >> 5, lane = tid & 31;
  const int qr0  = qblk * 64 + wave * 16;

  // async-stage 64 query rows (32 KB) into LDS
  {
    const _Float16* qsrc = qf + (((size_t)b * H_ + h) * S_ + qblk * 64) * HD_;
    for (int i = tid; i < 64 * HD_ / 8; i += 128)
      async_b128(lds_addr(&Qs[i * 8]), qsrc + (size_t)i * 8);
  }

  float8 ctx[16] = {};
  float m_i[8], l_i[8];
#pragma unroll
  for (int j = 0; j < 8; ++j) { m_i[j] = -INFINITY; l_i[j] = 0.f; }

  const _Float16* kptr = kf + (size_t)b * S_ * HD_;
  const _Float16* vptr = vf + (size_t)b * S_ * HD_;
  const int nlo = lane & 15, hh = lane >> 4;
  const int nblocks = (qblk * 64 + 64) / 32;     // causal upper bound

  for (int kb = 0; kb < nblocks; ++kb) {
    const int key0 = kb * 32;
    __syncthreads();   // previous block's LDS consumers done

    // prefetch next block of K/V into L2 while this one is being consumed
    if (kb + 1 < nblocks) {
      __builtin_prefetch(kptr + (size_t)(key0 + 32) * HD_ + tid * 64, 0, 1);
      __builtin_prefetch(vptr + (size_t)(key0 + 32) * HD_ + tid * 64, 0, 1);
    }

    // K block: async b128 copies (row-major)
    for (int i = tid; i < 32 * HD_ / 8; i += 128)
      async_b128(lds_addr(&Ks[i * 8]), kptr + (size_t)key0 * HD_ + i * 8);

    // V block: transpose while staging
    for (int i = tid; i < 32 * HD_ / 2; i += 128) {
      int key = (2 * i) >> 8;          // / HD_
      int d   = (2 * i) & (HD_ - 1);
      uint32_t vw = *(const uint32_t*)(vptr + (size_t)(key0 + key) * HD_ + d);
      Vt[(d + 0) * 32 + key] = ((const _Float16*)&vw)[0];
      Vt[(d + 1) * 32 + key] = ((const _Float16*)&vw)[1];
    }
    wait_async0();        // per-wave async completion (covers Q on kb==0)
    __syncthreads();

    // scores: two 16x16 tiles (keys 0..15, 16..31), K-dim = 256 in 8 WMMAs each
    float8 sA = {}, sB = {};
#pragma unroll
    for (int cc = 0; cc < 8; ++cc) {
      half16 a = frag_a(Qs, wave * 16, HD_, cc * 32, lane);
      sA = wmma_f16(a, frag_bT(Ks, 0,  HD_, cc * 32, lane), sA);
      sB = wmma_f16(a, frag_bT(Ks, 16, HD_, cc * 32, lane), sB);
    }

    // online softmax (per row j; rows j+8*hh)
    const int keyA = key0 + nlo, keyB = key0 + 16 + nlo;
    float alpha[8];
#pragma unroll
    for (int j = 0; j < 8; ++j) {
      const int qrow = qr0 + j + 8 * hh;
      float v0 = (keyA <= qrow) ? sA[j] : -INFINITY;
      float v1 = (keyB <= qrow) ? sB[j] : -INFINITY;
      float rm = fmaxf(v0, v1);
      rm = fmaxf(rm, __shfl_xor(rm, 1));
      rm = fmaxf(rm, __shfl_xor(rm, 2));
      rm = fmaxf(rm, __shfl_xor(rm, 4));
      rm = fmaxf(rm, __shfl_xor(rm, 8));
      const float mnew = fmaxf(m_i[j], rm);
      alpha[j] = __expf(m_i[j] - mnew);
      m_i[j] = mnew;
      float p0 = __expf(v0 - mnew), p1 = __expf(v1 - mnew);
      float rs = p0 + p1;
      rs += __shfl_xor(rs, 1);
      rs += __shfl_xor(rs, 2);
      rs += __shfl_xor(rs, 4);
      rs += __shfl_xor(rs, 8);
      l_i[j] = l_i[j] * alpha[j] + rs;
      sA[j] = p0;
      sB[j] = p1;
    }

    // stash P as f16 into per-wave LDS scratch (wave-local: DS ops in-order)
#pragma unroll
    for (int j = 0; j < 8; ++j) {
      const int row = j + 8 * hh;
      ps[wave][row * 32 + nlo]      = (_Float16)sA[j];
      ps[wave][row * 32 + 16 + nlo] = (_Float16)sB[j];
    }

    // ctx = ctx*alpha + P(16x32) @ V(32x256)
    half16 pa = frag_a(&ps[wave][0], 0, 32, 0, lane);
#pragma unroll
    for (int t = 0; t < 16; ++t) {
#pragma unroll
      for (int j = 0; j < 8; ++j) ctx[t][j] *= alpha[j];
      ctx[t] = wmma_f16(pa, frag_bT(Vt, t * 16, 32, 0, lane), ctx[t]);
    }
  }

  // normalize, restage into Qs (each wave owns its own 16 rows), vector store
  float rl[8];
#pragma unroll
  for (int j = 0; j < 8; ++j) rl[j] = 1.f / l_i[j];
#pragma unroll
  for (int t = 0; t < 16; ++t)
#pragma unroll
    for (int j = 0; j < 8; ++j)
      Qs[(wave * 16 + j + 8 * hh) * HD_ + t * 16 + nlo] =
          (_Float16)(ctx[t][j] * rl[j]);
  __syncthreads();

  // ctxf rows: [(b*S + s)*DOUT + h*HD + d], 256-half contiguous per row
  {
    _Float16* cbase = ctxf + ((size_t)b * S_ + qblk * 64) * DOUT + h * HD_;
    for (int i = tid; i < 64 * (HD_ / 8); i += 128) {
      int r = i >> 5;              // HD_/8 = 32 chunks per row
      int cidx = (i & 31) * 8;
      *(uint4*)(cbase + (size_t)r * DOUT + cidx) = *(const uint4*)(&Qs[r * HD_ + cidx]);
    }
  }
}

// ---------------------------------------------------------------------------
// Kernel 3: out = ctx @ Wo^T (f16 weights). grid: B*S/16, block 128.
// ---------------------------------------------------------------------------
__global__ __launch_bounds__(128)
void out_proj_kernel(const _Float16* __restrict__ ctxf,
                     const _Float16* __restrict__ Wo16,
                     float* __restrict__ out) {
  __shared__ _Float16 cs[16 * DOUT];     // 32 KB ctx tile
  __shared__ float    osf[16 * DIN];     // 40 KB f32 staging
  const int tile = blockIdx.x;
  const int tid  = threadIdx.x, wave = tid >> 5, lane = tid & 31;
  const int b    = tile / (S_ / 16);
  const int s0   = (tile % (S_ / 16)) * 16;

  {
    const _Float16* src = ctxf + ((size_t)b * S_ + s0) * DOUT;
    for (int i = tid; i < 16 * DOUT / 8; i += 128)
      async_b128(lds_addr(&cs[i * 8]), src + (size_t)i * 8);
  }
  wait_async0();
  __syncthreads();

  float8 c[10] = {};
  for (int kb = 0; kb < DOUT; kb += 32) {
    half16 a = frag_a(cs, 0, DOUT, kb, lane);
#pragma unroll
    for (int t = 0; t < 10; ++t)
      c[t] = wmma_f16(a, frag_bT(Wo16, wave * 160 + t * 16, DOUT, kb, lane), c[t]);
  }

  const int nlo = lane & 15, hh = lane >> 4;
#pragma unroll
  for (int t = 0; t < 10; ++t)
#pragma unroll
    for (int j = 0; j < 8; ++j)
      osf[(j + 8 * hh) * DIN + wave * 160 + t * 16 + nlo] = c[t][j];
  __syncthreads();

  // contiguous 16x640 f32 block store
  {
    float* op = out + ((size_t)b * S_ + s0) * DIN;
    const uint4* s = (const uint4*)osf;
    uint4* d = (uint4*)op;
    for (int i = tid; i < 16 * DIN / 4; i += 128) d[i] = s[i];
  }
}

// ---------------------------------------------------------------------------
extern "C" void kernel_launch(void* const* d_in, const int* in_sizes, int n_in,
                              void* d_out, int out_size, void* d_ws, size_t ws_size,
                              hipStream_t stream) {
  const float* x       = (const float*)d_in[0];
  const float* Wq      = (const float*)d_in[1];
  const float* Wk      = (const float*)d_in[2];
  const float* Wv      = (const float*)d_in[3];
  const float* Wo      = (const float*)d_in[4];
  const float* q_scale = (const float*)d_in[5];
  const float* k_scale = (const float*)d_in[6];
  const float* cosd    = (const float*)d_in[7];
  const float* sind    = (const float*)d_in[8];
  // d_in[9] = mask: causal, computed analytically in-kernel

  // f16 scratch layout in d_ws
  char* ws = (char*)d_ws;
  _Float16* qf   = (_Float16*)ws;                               // B*H*S*HD
  _Float16* kf   = qf + (size_t)B_ * H_ * S_ * HD_;             // B*S*HD
  _Float16* vf   = kf + (size_t)B_ * S_ * HD_;                  // B*S*HD
  _Float16* ctxf = vf + (size_t)B_ * S_ * HD_;                  // B*S*DOUT
  _Float16* wq16 = ctxf + (size_t)B_ * S_ * DOUT;               // DOUT*DIN
  _Float16* wk16 = wq16 + (size_t)DOUT * DIN;                   // HD*DIN
  _Float16* wv16 = wk16 + (size_t)HD_ * DIN;                    // HD*DIN
  _Float16* wo16 = wv16 + (size_t)HD_ * DIN;                    // DIN*DOUT
  float* out = (float*)d_out;

  // one-shot weight conversion (tiny vs. the matmuls)
  {
    int n2q = DOUT * DIN / 2, n2k = HD_ * DIN / 2, n2o = DIN * DOUT / 2;
    cvt_f32_f16_kernel<<<(n2q + 255) / 256, 256, 0, stream>>>(Wq, wq16, n2q);
    cvt_f32_f16_kernel<<<(n2k + 255) / 256, 256, 0, stream>>>(Wk, wk16, n2k);
    cvt_f32_f16_kernel<<<(n2k + 255) / 256, 256, 0, stream>>>(Wv, wv16, n2k);
    cvt_f32_f16_kernel<<<(n2o + 255) / 256, 256, 0, stream>>>(Wo, wo16, n2o);
  }

  dim3 g1(B_ * S_ / 16, H_ + 2);
  qkv_norm_rope_kernel<<<g1, 128, 0, stream>>>(x, wq16, wk16, wv16,
                                               q_scale, k_scale, cosd, sind,
                                               qf, kf, vf);

  dim3 g2(S_ / 64, B_ * H_);
  flash_attn_kernel<<<g2, 128, 0, stream>>>(qf, kf, vf, ctxf);

  out_proj_kernel<<<B_ * S_ / 16, 128, 0, stream>>>(ctxf, wo16, out);
}
